// MyGRU_77171972374841
// MI455X (gfx1250) — matrix operand order repaired and checked by reading, not compiled, not run on hardware
//
#include <hip/hip_runtime.h>
#include <hip/hip_bf16.h>
#include <stdint.h>

// ---------------- problem constants ----------------
#define B_   32
#define S_   1024
#define IN_  512
#define H_   1024
#define KC_  (IN_ + H_)        // 1536

// sequential-phase config: 32 WGs x 128 threads (4 waves), 32 H-channels/WG
#define NWG_SEQ     32
#define CHUNK       32
#define SEQ_THREADS 128
#define LDS_W_ELEMS (3 * CHUNK * H_)       // 98304 bf16 = 192 KB (weights, resident all steps)
#define LDS_A_ELEMS (B_ * H_)              // 32768 bf16 = 64 KB (h / r*h snapshot)
#define LDS_BYTES   ((LDS_W_ELEMS + LDS_A_ELEMS) * 2)   // 256 KB dynamic LDS (<= 320 KB/WGP)

typedef __bf16 bf16;
typedef __attribute__((ext_vector_type(16))) __bf16 v16bf;
typedef __attribute__((ext_vector_type(8)))  float  v8f;

union FragBF { v16bf v; uint4 q[2]; };
union FragF  { v8f   v; float f[8]; };

#define WMMA_BF16(a, b, c) \
  __builtin_amdgcn_wmma_f32_16x16x32_bf16(false, (a), false, (b), (short)0, (c), false, false)

// ---- CDNA5 async global->LDS copy (ASYNCcnt-tracked, no VGPR round trip) ----
// LDS destination operand = wave-relative LDS byte offset; for a generic pointer
// into __shared__ the flat address low 32 bits are exactly that offset (ISA 10.2).
__device__ __forceinline__ void async_copy_b128(void* lds_dst, const void* gsrc) {
  unsigned loff = (unsigned)(uintptr_t)lds_dst;
  asm volatile("global_load_async_to_lds_b128 %0, %1, off"
               :: "v"(loff), "v"(gsrc) : "memory");
}
__device__ __forceinline__ void wait_async0() {
  asm volatile("s_wait_asynccnt 0x0" ::: "memory");
}

// A fragment (16x32, bf16): lane<16 -> row=lane, K {k0+0..7, k0+16..23}; lane>=16 -> +8 K-offset
__device__ __forceinline__ v16bf load_a(const bf16* __restrict__ base, int ld, int row0, int k0) {
  const int lane = threadIdx.x & 31;
  const bf16* p = base + (size_t)(row0 + (lane & 15)) * ld + k0 + ((lane & 16) ? 8 : 0);
  FragBF f;
  f.q[0] = *(const uint4*)(p);
  f.q[1] = *(const uint4*)(p + 16);
  return f.v;
}

// B fragment (32x16, bf16) from row-major W[n][k]: lane<16 -> col n=lane, K {k0+0..7, k0+16..23};
// lane>=16 -> col n=lane-16, +8 K-offset (matches load_a)
__device__ __forceinline__ v16bf load_b(const bf16* __restrict__ base, int ld, int row0, int k0) {
  const int lane = threadIdx.x & 31;
  const bf16* p = base + (size_t)(row0 + (lane & 15)) * ld + k0 + ((lane & 16) ? 8 : 0);
  FragBF f;
  f.q[0] = *(const uint4*)(p);
  f.q[1] = *(const uint4*)(p + 16);
  return f.v;
}

__device__ __forceinline__ float fast_sigmoid(float x) {
  return 1.0f / (1.0f + __expf(-x));
}
__device__ __forceinline__ float fast_tanh(float x) {
  float e = __expf(2.0f * x);
  return (e - 1.0f) / (e + 1.0f);
}

// Device-wide barrier: monotonically increasing counter in ws (zeroed each launch by gru_init).
__device__ __forceinline__ void grid_barrier(unsigned* cnt, unsigned target) {
  __threadfence();            // release: flush our global stores device-wide
  __syncthreads();
  if (threadIdx.x == 0) {
    __hip_atomic_fetch_add(cnt, 1u, __ATOMIC_ACQ_REL, __HIP_MEMORY_SCOPE_AGENT);
    while (__hip_atomic_load(cnt, __ATOMIC_ACQUIRE, __HIP_MEMORY_SCOPE_AGENT) < target) {
      __builtin_amdgcn_s_sleep(2);
    }
  }
  __syncthreads();
  __threadfence();            // acquire: invalidate stale WGP-cache lines
}

// ---------------- kernel 1: pack all three weight matrices fp32 -> bf16 ----------------
__global__ void gru_pack_w(const float* __restrict__ Wz, const float* __restrict__ Wr,
                           const float* __restrict__ Wh, bf16* __restrict__ Wp) {
  size_t i = (size_t)blockIdx.x * blockDim.x + threadIdx.x;
  const size_t n1 = (size_t)H_ * KC_;
  if (i < n1)            Wp[i] = (bf16)Wz[i];
  else if (i < 2 * n1)   Wp[i] = (bf16)Wr[i - n1];
  else if (i < 3 * n1)   Wp[i] = (bf16)Wh[i - 2 * n1];
}

// ---------------- kernel 2: init h state (fp32 + bf16 copies) and barrier counters ----------------
__global__ void gru_init(const float* __restrict__ h0, float* __restrict__ hf,
                         bf16* __restrict__ hb, unsigned* __restrict__ bar) {
  int i = blockIdx.x * blockDim.x + threadIdx.x;
  if (i < B_ * H_) {
    float v = h0[i % H_];           // broadcast [1,H] over batch
    hf[i] = v;
    hb[i] = (bf16)v;
  }
  if (i < 64) bar[i] = 0u;
}

// ---------------- kernel 3: parallel precompute Gx[g][s][b][n] = x_s @ Wx_g^T + bias_g ----------------
// One WG per (s, gate). bf16 WMMA, K=512, M=32 (batch), N=1024.
__global__ __launch_bounds__(256) void gru_gatex(
    const float* __restrict__ x,     // [B][S][IN] fp32
    const bf16*  __restrict__ Wp,    // [3][H][KC] bf16 (cols 0..511 = x-part)
    const float* __restrict__ bz, const float* __restrict__ br, const float* __restrict__ bh,
    bf16* __restrict__ Gx)           // [3][S][B][H] bf16
{
  __shared__ bf16 lds_x[B_ * IN_];   // 32 KB: x[:, s, :] as bf16
  const int s   = blockIdx.x;
  const int g   = blockIdx.y;
  const int tid = threadIdx.x;

  // stage + convert x slice: 32 rows x 512 fp32 -> bf16
  for (int i = tid; i < (B_ * IN_) / 4; i += 256) {
    int b = i / (IN_ / 4), c = i % (IN_ / 4);
    float4 xv = *(const float4*)(x + ((size_t)b * S_ + s) * IN_ + c * 4);
    bf16* d = lds_x + b * IN_ + c * 4;
    d[0] = (bf16)xv.x; d[1] = (bf16)xv.y; d[2] = (bf16)xv.z; d[3] = (bf16)xv.w;
  }
  __syncthreads();

  const int wave = tid >> 5, lane = tid & 31;
  const float* bias = (g == 0) ? bz : (g == 1) ? br : bh;
  const bf16* Wg = Wp + (size_t)g * H_ * KC_;
  bf16* outg = Gx + (((size_t)g * S_ + s) * B_) * H_;

  for (int t = wave; t < 128; t += 8) {        // 2 m-tiles x 64 n-tiles
    const int mt = t & 1, nt = t >> 1;
    v8f acc0 = {}, acc1 = {};
    #pragma unroll
    for (int ks = 0; ks < 16; ks += 2) {       // K = 512 = 16 x 32
      v16bf a0 = load_a(lds_x, IN_, mt * 16, ks * 32);
      v16bf w0 = load_b(Wg,    KC_, nt * 16, ks * 32);
      acc0 = WMMA_BF16(a0, w0, acc0);
      v16bf a1 = load_a(lds_x, IN_, mt * 16, (ks + 1) * 32);
      v16bf w1 = load_b(Wg,    KC_, nt * 16, (ks + 1) * 32);
      acc1 = WMMA_BF16(a1, w1, acc1);
    }
    FragF acc; acc.v = acc0 + acc1;
    const int n = nt * 16 + (lane & 15);
    const float bsv = bias[n];
    #pragma unroll
    for (int v = 0; v < 8; v++) {
      int bi = mt * 16 + v + ((lane & 16) ? 8 : 0);
      outg[(size_t)bi * H_ + n] = (bf16)(acc.f[v] + bsv);
    }
  }
}

// ---------------- kernel 4: sequential scan over S with LDS-resident recurrent weights ----------------
__global__ __launch_bounds__(SEQ_THREADS, 1) void gru_seq(
    const bf16* __restrict__ Wp,    // [3][H][KC] (cols 512..1535 = recurrent part)
    const bf16* __restrict__ Gx,    // [3][S][B][H]
    float* __restrict__ hf,         // [B][H] fp32 state
    bf16*  __restrict__ hb,         // [B][H] bf16 state (GEMM operand)
    bf16*  __restrict__ rhb,        // [B][H] bf16 r*h
    unsigned* __restrict__ bar,
    float* __restrict__ out)        // [B][S][H] then final [B][H]
{
  extern __shared__ bf16 lds[];
  bf16* lds_w = lds;                         // [3][CHUNK][H] recurrent weight slice
  bf16* lds_a = lds + LDS_W_ELEMS;           // [B][H] h or r*h snapshot

  const int tid  = threadIdx.x;
  const int lane = tid & 31;
  const int wave = tid >> 5;                 // 4 waves: (mt, nt) in 2x2
  const int mt = wave & 1, nt = wave >> 1;
  const int n0g = blockIdx.x * CHUNK;        // this WG's global channel base

  // prologue: recurrent weight slice global -> LDS via async DMA (once; resident all steps)
  for (int i = tid; i < LDS_W_ELEMS / 8; i += SEQ_THREADS) {
    int row = i / (H_ / 8), c8 = i % (H_ / 8);        // row in [0,96): g*32 + n_local
    int g = row / CHUNK, nl = row % CHUNK;
    async_copy_b128(lds_w + (size_t)row * H_ + c8 * 8,
                    Wp + ((size_t)g * H_ + n0g + nl) * KC_ + IN_ + c8 * 8);
  }
  wait_async0();
  __syncthreads();

  const int m_row = mt * 16;
  const int nl0   = nt * 16;
  const int n_col = n0g + nl0 + (lane & 15);
  const size_t GS = (size_t)S_ * B_ * H_;

  for (int s = 0; s < S_; s++) {
    grid_barrier(bar, (unsigned)(2 * s + 1) * NWG_SEQ);   // h from step s-1 visible

    // stage h snapshot into LDS via async DMA (full [B][H])
    for (int i = tid; i < LDS_A_ELEMS / 8; i += SEQ_THREADS)
      async_copy_b128(lds_a + i * 8, hb + i * 8);

    // prefetch next step's Gx slices (streamed from HBM) while DMA runs
    if (s + 1 < S_) {
      #pragma unroll
      for (int g = 0; g < 3; g++) {
        const bf16* p = Gx + (size_t)g * GS + (((size_t)(s + 1)) * B_) * H_;
        __builtin_prefetch((const void*)(p + (size_t)tid * 256), 0, 1);
      }
    }
    wait_async0();
    __syncthreads();

    // ---- phase 1: z and r for this wave's 16x16 tile (K = H = 1024) ----
    v8f az = {}, ar = {};
    for (int ks = 0; ks < 32; ks++) {
      v16bf a  = load_a(lds_a, H_, m_row, ks * 32);
      v16bf wz = load_b(lds_w + 0 * CHUNK * H_, H_, nl0, ks * 32);
      v16bf wr = load_b(lds_w + 1 * CHUNK * H_, H_, nl0, ks * 32);
      az = WMMA_BF16(a, wz, az);
      ar = WMMA_BF16(a, wr, ar);
    }
    FragF faz, far_; faz.v = az; far_.v = ar;
    float zv[8], hpv[8];
    #pragma unroll
    for (int v = 0; v < 8; v++) {
      int bi = m_row + v + ((lane & 16) ? 8 : 0);
      size_t go = ((size_t)s * B_ + bi) * H_ + n_col;
      float gz = (float)Gx[go];
      float gr = (float)Gx[GS + go];
      float hv = hf[(size_t)bi * H_ + n_col];
      float z  = fast_sigmoid(faz.f[v] + gz);
      float r  = fast_sigmoid(far_.f[v] + gr);
      rhb[(size_t)bi * H_ + n_col] = (bf16)(r * hv);
      zv[v] = z; hpv[v] = hv;
    }

    grid_barrier(bar, (unsigned)(2 * s + 2) * NWG_SEQ);   // full r*h visible

    // stage r*h snapshot into LDS via async DMA
    for (int i = tid; i < LDS_A_ELEMS / 8; i += SEQ_THREADS)
      async_copy_b128(lds_a + i * 8, rhb + i * 8);
    wait_async0();
    __syncthreads();

    // ---- phase 2: h_hat GEMM, then blend ----
    v8f ah0 = {}, ah1 = {};
    for (int ks = 0; ks < 32; ks += 2) {
      v16bf a0 = load_a(lds_a, H_, m_row, ks * 32);
      v16bf w0 = load_b(lds_w + 2 * CHUNK * H_, H_, nl0, ks * 32);
      ah0 = WMMA_BF16(a0, w0, ah0);
      v16bf a1 = load_a(lds_a, H_, m_row, (ks + 1) * 32);
      v16bf w1 = load_b(lds_w + 2 * CHUNK * H_, H_, nl0, (ks + 1) * 32);
      ah1 = WMMA_BF16(a1, w1, ah1);
    }
    FragF fah; fah.v = ah0 + ah1;
    #pragma unroll
    for (int v = 0; v < 8; v++) {
      int bi = m_row + v + ((lane & 16) ? 8 : 0);
      size_t go = ((size_t)s * B_ + bi) * H_ + n_col;
      float gh = (float)Gx[2 * GS + go];
      float hh = fast_tanh(fah.f[v] + gh);
      float hn = hpv[v] + zv[v] * (hh - hpv[v]);       // (1-z)*h + z*h_hat
      out[((size_t)bi * S_ + s) * H_ + n_col] = hn;
      hf[(size_t)bi * H_ + n_col] = hn;
      hb[(size_t)bi * H_ + n_col] = (bf16)hn;
      if (s == S_ - 1)
        out[(size_t)B_ * S_ * H_ + (size_t)bi * H_ + n_col] = hn;   // hidden_final
    }
  }
}

// ---------------- host launcher ----------------
extern "C" void kernel_launch(void* const* d_in, const int* in_sizes, int n_in,
                              void* d_out, int out_size, void* d_ws, size_t ws_size,
                              hipStream_t stream) {
  const float* x  = (const float*)d_in[0];
  const float* Wz = (const float*)d_in[1];
  const float* bz = (const float*)d_in[2];
  const float* Wr = (const float*)d_in[3];
  const float* br = (const float*)d_in[4];
  const float* Wh = (const float*)d_in[5];
  const float* bh = (const float*)d_in[6];
  const float* h0 = (const float*)d_in[7];

  // workspace layout (~211 MB total)
  char* ws = (char*)d_ws;
  const size_t offW  = 0;                                            // 9,437,184 B
  const size_t offG  = offW  + (size_t)3 * H_ * KC_ * sizeof(bf16);  // +201,326,592 B
  const size_t offHf = offG  + (size_t)3 * S_ * B_ * H_ * sizeof(bf16);
  const size_t offHb = offHf + (size_t)B_ * H_ * sizeof(float);
  const size_t offRh = offHb + (size_t)B_ * H_ * sizeof(bf16);
  const size_t offBar= offRh + (size_t)B_ * H_ * sizeof(bf16);

  bf16*     Wp  = (bf16*)(ws + offW);
  bf16*     Gx  = (bf16*)(ws + offG);
  float*    hf  = (float*)(ws + offHf);
  bf16*     hb  = (bf16*)(ws + offHb);
  bf16*     rhb = (bf16*)(ws + offRh);
  unsigned* bar = (unsigned*)(ws + offBar);

  // 1. pack weights fp32 -> bf16  (3 * 1024 * 1536 elements)
  gru_pack_w<<<(3 * H_ * KC_ + 255) / 256, 256, 0, stream>>>(Wz, Wr, Wh, Wp);
  // 2. init h state + zero barrier counters (re-zeroed every launch -> graph-replay safe)
  gru_init<<<(B_ * H_ + 255) / 256, 256, 0, stream>>>(h0, hf, hb, bar);
  // 3. parallel x-part precompute: one WG per (timestep, gate)
  gru_gatex<<<dim3(S_, 3), 256, 0, stream>>>(x, Wp, bz, br, bh, Gx);
  // 4. sequential scan: 32 co-resident WGs, 256 KB LDS each (weights stay resident)
  (void)hipFuncSetAttribute((const void*)gru_seq,
                            hipFuncAttributeMaxDynamicSharedMemorySize, LDS_BYTES);
  gru_seq<<<NWG_SEQ, SEQ_THREADS, LDS_BYTES, stream>>>(Wp, Gx, hf, hb, rhb, bar,
                                                       (float*)d_out);
}
